// GroupConv2D_23596550324377
// MI455X (gfx1250) — matrix operand order repaired and checked by reading, not compiled, not run on hardware
//
#include <hip/hip_runtime.h>

typedef __attribute__((ext_vector_type(16))) __bf16       v16bf;
typedef __attribute__((ext_vector_type(8)))  __bf16       v8bf;
typedef __attribute__((ext_vector_type(4)))  __bf16       v4bf;
typedef __attribute__((ext_vector_type(8)))  float        v8f;
typedef __attribute__((ext_vector_type(4)))  float        v4f;
typedef __attribute__((ext_vector_type(4)))  unsigned int u32x4;
typedef __attribute__((ext_vector_type(8)))  int          i32x8;
typedef __attribute__((ext_vector_type(4)))  int          i32x4;

#define BATCH  32
#define HSZ    56
#define WSZ    56
#define CIN    256
#define COUT   256
#define NGRP   8
#define CG     32                        // channels per group
#define TAPS   9

#define HROWS  4                         // output rows per block
#define TROWS  6                         // staged input rows (with halo)
#define WPAD   58                        // 56 + 2 zero pad columns
#define PIXSH  40                        // shorts per pixel in LDS (80B, bank-spread)
#define ROWSH  (WPAD * PIXSH)            // 2320 shorts per staged row
#define BPX    (HROWS * WSZ)             // 224 pixels per block
#define NTHR   224                       // 7 waves

#define RAW_FLOATS  (TAPS * CG * CG)     // 9216 fp32  = 36 KB staged by TDM
#define FRAG_SHORTS (TAPS * 2 * 32 * 16) // 9216 bf16 = 18 KB B-fragments

#if __has_builtin(__builtin_amdgcn_tensor_load_to_lds)
#define HAVE_TDM 1
#else
#define HAVE_TDM 0
#endif

__device__ __forceinline__ int clampi(int v, int lo, int hi) {
    return v < lo ? lo : (v > hi ? hi : v);
}

__device__ __forceinline__ v8f wmma_bf16(v16bf a, v16bf b, v8f c) {
    return __builtin_amdgcn_wmma_f32_16x16x32_bf16(
        false, a, false, b, (short)0, c, false, false);
}

// Block: one image x 4 output rows x one group (224 px, 7 waves).
// Wave: 32 px (2 x 16-row A tiles) x 32 out channels; 36 WMMAs, fp32 accum.
__global__ void __launch_bounds__(NTHR)
GroupConv2D_23596550324377_kernel(const float* __restrict__ in,
                                  const float* __restrict__ wt,
                                  const float* __restrict__ bias,
                                  float* __restrict__ out)
{
#if HAVE_TDM
    __shared__ __align__(128) float wraw[RAW_FLOATS];           // 36 KB via TDM
#endif
    __shared__ __align__(32) unsigned short wfr16[FRAG_SHORTS]; // 18 KB B frags
    __shared__ __align__(16) unsigned short tlds[TROWS * ROWSH]; // 27.2 KB A tile

    const int g   = blockIdx.y;
    const int bx  = blockIdx.x;              // 0..447
    const int b   = bx / (HSZ / HROWS);      // image
    const int h0  = (bx - b * (HSZ / HROWS)) * HROWS; // first output row
    const int tid = threadIdx.x;

#if HAVE_TDM
    // ---- TDM: DMA this group's weights [288 rows x 32 floats, stride 256] ----
    if (tid == 0) {
        unsigned long long ga = (unsigned long long)(const void*)(wt + g * CG);
        unsigned lds_off = (unsigned)(unsigned long long)
            (__attribute__((address_space(3))) void*)wraw;
        u32x4 d0;
        d0[0] = 1u;                                   // count=1, user descriptor
        d0[1] = lds_off;                              // lds_addr (bytes)
        d0[2] = (unsigned)ga;                         // global_addr[31:0]
        d0[3] = ((unsigned)(ga >> 32) & 0x01FFFFFFu)  // global_addr[56:32]
              | (2u << 30);                           // type = 2 ("image")
        i32x8 d1;
        d1[0] = (int)(2u << 16);                      // data_size = 2 -> 4 bytes
        d1[1] = (int)(CG << 16);                      // tensor_dim0 = 32 (lo16)
        d1[2] = (int)((TAPS * CG) << 16);             // tensor_dim1 = 288 (lo16)
        d1[3] = (int)(CG << 16);                      // tile_dim0 = 32
        d1[4] = (int)(TAPS * CG);                     // tile_dim1 = 288
        d1[5] = (int)COUT;                            // tensor_dim0_stride = 256
        d1[6] = 0;
        d1[7] = 0;
        i32x4 dz = {0, 0, 0, 0};
#if __clang_major__ >= 23
        i32x8 dz8 = {0, 0, 0, 0, 0, 0, 0, 0};
        __builtin_amdgcn_tensor_load_to_lds(d0, d1, dz, dz, dz8, 0);
#else
        __builtin_amdgcn_tensor_load_to_lds(d0, d1, dz, dz, 0);
#endif
#if __has_builtin(__builtin_amdgcn_s_wait_tensorcnt)
        __builtin_amdgcn_s_wait_tensorcnt(0);
#else
        asm volatile("s_wait_tensorcnt 0x0" ::: "memory");
#endif
    }
    __syncthreads();
#endif

    // ---- Weights: convert to bf16 B-fragment layout in LDS ----
    // B 32x16 bf16 order used here, same k as A: lane = N + 16*kg ; slot s -> K = kg*16 + s
    for (int e = tid; e < RAW_FLOATS; e += NTHR) {
        const int tap = e >> 10;
        const int rem = e & 1023;
        const int ci  = rem >> 5;
        const int co  = rem & 31;
#if HAVE_TDM
        const float w = wraw[e];
#else
        const float w = wt[(tap * CG + ci) * COUT + g * CG + co];
#endif
        const int frag = tap * 2 + (co >> 4);
        const int ln   = (co & 15) | (((ci >> 4) & 1) << 4);
        const int s    = ci & 15;
        wfr16[frag * 512 + ln * 16 + s] =
            __builtin_bit_cast(unsigned short, (__bf16)w);
    }

    // ---- Input: stage 6 halo rows x 56 w x 32 ch as bf16, converted ONCE ----
    // Channel slots in natural order [0..31] (contiguous k = 16*kg + s) so a
    // lane's 16-ch A chunk is one contiguous 32B LDS read.
    for (int c = tid; c < TROWS * WSZ * 8; c += NTHR) {   // 2688 4-float chunks
        const int r  = c / (WSZ * 8);
        const int rm = c - r * (WSZ * 8);
        const int w  = rm >> 3;
        const int cc = rm & 7;                // 4-channel chunk index
        const int h  = h0 + r - 1;
        const bool hv = (h >= 0) & (h < HSZ);
        const int hc = clampi(h, 0, HSZ - 1);
        const v4f rr = *(const v4f*)
            (in + ((b * HSZ + hc) * WSZ + w) * CIN + g * CG + cc * 4);
        v4bf pk;
        #pragma unroll
        for (int i = 0; i < 4; ++i) pk[i] = (__bf16)rr[i];
        const v4bf z = {};
        pk = hv ? pk : z;                     // zero OOB halo rows
        const int bi = cc >> 1;               // 8-ch block 0..3
        const int sb = bi;                    // natural block order
        *(v4bf*)(tlds + r * ROWSH + (w + 1) * PIXSH + sb * 8 + (cc & 1) * 4) = pk;
    }

    // ---- Zero the two pad columns (w = -1 and w = 56) ----
    for (int zz = tid; zz < TROWS * 2 * (PIXSH / 2); zz += NTHR) { // 240 dwords
        const int r   = zz / (2 * (PIXSH / 2));
        const int k   = zz - r * (2 * (PIXSH / 2));
        const int col = k / (PIXSH / 2);
        const int q   = k - col * (PIXSH / 2);
        ((unsigned*)tlds)[r * (ROWSH / 2) + (col ? 57 * (PIXSH / 2) : 0) + q] = 0u;
    }
    __syncthreads();

    // ---- Main loop: all halo logic is gone; tap offsets are DS immediates ----
    const int lane = tid & 31;
    const int wv   = tid >> 5;
    const int m    = lane & 15;
    const int kg   = lane >> 4;

    const int p0  = wv * 32 + m;              // block-local pixel, A tile 0
    const int p1  = p0 + 16;                  // A tile 1
    const int hl0 = p0 / WSZ, wl0 = p0 - hl0 * WSZ;
    const int hl1 = p1 / WSZ, wl1 = p1 - hl1 * WSZ;
    const unsigned baseA0 = hl0 * ROWSH + (wl0 + 1) * PIXSH + kg * 16;
    const unsigned baseA1 = hl1 * ROWSH + (wl1 + 1) * PIXSH + kg * 16;

    const v16bf* __restrict__ wfr = (const v16bf*)wfr16;

    v8f acc00 = {}, acc01 = {}, acc10 = {}, acc11 = {};

    #pragma unroll
    for (int tap = 0; tap < TAPS; ++tap) {
        const int kh = tap / 3;
        const int kw = tap - 3 * kh;
        const unsigned toff = kh * ROWSH + kw * PIXSH;  // constant per tap

        const v8bf a0lo = *(const v8bf*)(tlds + baseA0 + toff);
        const v8bf a0hi = *(const v8bf*)(tlds + baseA0 + toff + 8);
        const v8bf a1lo = *(const v8bf*)(tlds + baseA1 + toff);
        const v8bf a1hi = *(const v8bf*)(tlds + baseA1 + toff + 8);
        const v16bf a0 = __builtin_shufflevector(a0lo, a0hi,
            0,1,2,3,4,5,6,7,8,9,10,11,12,13,14,15);
        const v16bf a1 = __builtin_shufflevector(a1lo, a1hi,
            0,1,2,3,4,5,6,7,8,9,10,11,12,13,14,15);

        const v16bf wb0 = wfr[(tap * 2 + 0) * 32 + lane];
        const v16bf wb1 = wfr[(tap * 2 + 1) * 32 + lane];

        acc00 = wmma_bf16(a0, wb0, acc00);
        acc01 = wmma_bf16(a0, wb1, acc01);
        acc10 = wmma_bf16(a1, wb0, acc10);
        acc11 = wmma_bf16(a1, wb1, acc11);
    }

    // ---- Epilogue: block pixels are contiguous in flat NHWC; NT stores ----
    const int n0 = lane & 15;
    const float bia0 = bias[g * CG + n0];
    const float bia1 = bias[g * CG + 16 + n0];
    const unsigned pixbase = (unsigned)((b * HSZ + h0) * WSZ) + wv * 32;
    const unsigned obase   = pixbase * COUT + g * CG + n0;

    #pragma unroll
    for (int r = 0; r < 8; ++r) {
        const unsigned oA = obase + (unsigned)(r + 8 * kg) * COUT;
        const unsigned oB = oA + 16u * COUT;
        __builtin_nontemporal_store(acc00[r] + bia0, out + oA);
        __builtin_nontemporal_store(acc01[r] + bia1, out + oA + 16);
        __builtin_nontemporal_store(acc10[r] + bia0, out + oB);
        __builtin_nontemporal_store(acc11[r] + bia1, out + oB + 16);
    }
}

extern "C" void kernel_launch(void* const* d_in, const int* in_sizes, int n_in,
                              void* d_out, int out_size, void* d_ws, size_t ws_size,
                              hipStream_t stream) {
    (void)in_sizes; (void)n_in; (void)out_size; (void)d_ws; (void)ws_size;
    const float* in   = (const float*)d_in[0];  // [32,56,56,256] NHWC
    const float* wt   = (const float*)d_in[1];  // [3,3,32,256]   HWIO
    const float* bias = (const float*)d_in[2];  // [256]
    float* out        = (float*)d_out;          // [32,56,56,256]

    dim3 grid(BATCH * (HSZ / HROWS), NGRP, 1);  // 448 x 8 blocks
    dim3 block(NTHR, 1, 1);                     // 7 waves of 32
    GroupConv2D_23596550324377_kernel<<<grid, block, 0, stream>>>(in, wt, bias, out);
}
